// GCNRouting2Hop_32229434589308
// MI455X (gfx1250) — matrix operand order repaired and checked by reading, not compiled, not run on hardware
//
#include <hip/hip_runtime.h>
#include <hip/hip_bf16.h>

// ---------------- types ----------------
typedef __bf16 bf16_t;
typedef __attribute__((ext_vector_type(16))) __bf16 v16bf;
typedef __attribute__((ext_vector_type(8)))  __bf16 v8bf;
typedef __attribute__((ext_vector_type(8)))  float  v8f;

#define DFEAT 128

__device__ __forceinline__ void atomic_add_f32(float* p, float v) {
    __hip_atomic_fetch_add(p, v, __ATOMIC_RELAXED, __HIP_MEMORY_SCOPE_AGENT);
}

// ---------------- elementwise converts ----------------
__global__ void k_cvt_bf16(const float* __restrict__ in, bf16_t* __restrict__ out, long n) {
    long i = (long)blockIdx.x * blockDim.x + threadIdx.x;
    if (i < n) out[i] = (bf16_t)in[i];
}

__global__ void k_relu_cvt_bf16(const float* __restrict__ in, bf16_t* __restrict__ out, long n) {
    long i = (long)blockIdx.x * blockDim.x + threadIdx.x;
    if (i < n) out[i] = (bf16_t)fmaxf(in[i], 0.0f);
}

// WT[n*128+k] = bf16(W[k*128+n]) for both layer weights (128x128 each)
__global__ void k_transpose_w(const float* __restrict__ W1, const float* __restrict__ W2,
                              bf16_t* __restrict__ W1T, bf16_t* __restrict__ W2T) {
    int idx = blockIdx.x * blockDim.x + threadIdx.x;   // 0..16383
    int n = idx >> 7, k = idx & (DFEAT - 1);
    W1T[idx] = (bf16_t)W1[k * DFEAT + n];
    W2T[idx] = (bf16_t)W2[k * DFEAT + n];
}

// ---------------- degree / norm ----------------
__global__ void k_fill_zero(float* __restrict__ p, int n) {
    int i = blockIdx.x * blockDim.x + threadIdx.x;
    if (i < n) p[i] = 0.0f;
}

__global__ void k_degree(const int* __restrict__ dst, float* __restrict__ deg, int E) {
    int e = blockIdx.x * blockDim.x + threadIdx.x;
    if (e < E) atomic_add_f32(&deg[dst[e]], 1.0f);
}

__global__ void k_dinv(float* __restrict__ deg, int N) {
    int i = blockIdx.x * blockDim.x + threadIdx.x;
    if (i < N) deg[i] = __frsqrt_rn(deg[i] + 1.0f);   // deg includes +1 self loop
}

// ---------------- WMMA GEMM:  C[N x 128] = A(bf16, N x 128) @ W(bf16, 128x128) ----------------
// BT is W transposed:  BT[n*128 + k] = W[k][n].
// One wave -> one 16x16 tile; block = 8 waves = all 8 column tiles of one row tile.
__global__ void k_gemm_wmma_bf16(const bf16_t* __restrict__ A,
                                 const bf16_t* __restrict__ BT,
                                 float* __restrict__ C) {
    const int lane  = threadIdx.x & 31;
    const int wave  = threadIdx.x >> 5;        // ntile: 0..7
    const int mtile = blockIdx.x;              // row tile
    const int l16   = lane & 15;
    const int hi    = lane >> 4;               // 0: lanes 0-15, 1: lanes 16-31

    const bf16_t* __restrict__ arow = A  + (long)(mtile * 16 + l16) * DFEAT;
    const bf16_t* __restrict__ brow = BT + (long)(wave  * 16 + l16) * DFEAT;

    v8f acc = {};
#pragma unroll
    for (int kk = 0; kk < 4; ++kk) {
        // A fragment (ISA 16-bit A 16x32 layout): this lane holds row M=l16,
        // K = 32*kk + 8*hi + {0..7}  (vgprs 0-3)  and  +16..+23 (vgprs 4-7)
        const int klo = kk * 32 + hi * 8;
        v8bf alo = *(const v8bf*)(arow + klo);
        v8bf ahi = *(const v8bf*)(arow + klo + 16);
        v16bf a;
#pragma unroll
        for (int i = 0; i < 8; ++i) { a[i] = alo[i]; a[8 + i] = ahi[i]; }

        // B fragment: this lane holds column N=l16, K = 32*kk + 8*hi + {0..7} and +16..+23
        const int kb = kk * 32 + hi * 8;
        v8bf blo = *(const v8bf*)(brow + kb);
        v8bf bhi = *(const v8bf*)(brow + kb + 16);
        v16bf b;
#pragma unroll
        for (int i = 0; i < 8; ++i) { b[i] = blo[i]; b[8 + i] = bhi[i]; }

        acc = __builtin_amdgcn_wmma_f32_16x16x32_bf16(
            /*neg_a=*/false, a, /*neg_b=*/false, b,
            /*c_mod=*/(short)0, acc, /*reuse_a=*/false, /*reuse_b=*/false);
    }

    // C layout: vgpr r holds M = r + 8*hi, column N = l16
    const int mbase = mtile * 16 + hi * 8;
    const int cbase = wave * 16 + l16;
#pragma unroll
    for (int r = 0; r < 8; ++r)
        C[(long)(mbase + r) * DFEAT + cbase] = acc[r];
}

// ---------------- aggregation ----------------
// agg[i,:] = h[i,:] * dinv[i]^2 + bias[:]     (self-loop term + bias folded into init)
__global__ void k_init_agg(const float* __restrict__ h, const float* __restrict__ dinv,
                           const float* __restrict__ bias, float* __restrict__ agg, long n) {
    long idx = (long)blockIdx.x * blockDim.x + threadIdx.x;
    if (idx < n) {
        int i = (int)(idx >> 7);
        int j = (int)(idx & (DFEAT - 1));
        float di = dinv[i];
        agg[idx] = h[idx] * di * di + bias[j];
    }
}

// one wave per edge: agg[dst,:] += h[src,:] * dinv[src]*dinv[dst]
__global__ void k_scatter(const float* __restrict__ h,
                          const int* __restrict__ src, const int* __restrict__ dst,
                          const float* __restrict__ dinv,
                          float* __restrict__ agg, int E) {
    int e    = blockIdx.x * (blockDim.x >> 5) + (threadIdx.x >> 5);
    int lane = threadIdx.x & 31;
    if (e >= E) return;
    int s = src[e], d = dst[e];
    float norm = dinv[s] * dinv[d];
    const float* __restrict__ hs = h   + (long)s * DFEAT;
    float*       __restrict__ ad = agg + (long)d * DFEAT;
    __builtin_prefetch(ad + lane, 1, 0);   // global_prefetch of destination row
#pragma unroll
    for (int j = 0; j < DFEAT; j += 32)
        atomic_add_f32(&ad[j + lane], hs[j + lane] * norm);
}

// ---------------- launch ----------------
extern "C" void kernel_launch(void* const* d_in, const int* in_sizes, int n_in,
                              void* d_out, int out_size, void* d_ws, size_t ws_size,
                              hipStream_t stream) {
    const float* x    = (const float*)d_in[0];
    const int*   eidx = (const int*)  d_in[1];   // (2, E) flattened
    const float* W1   = (const float*)d_in[2];
    const float* b1   = (const float*)d_in[3];
    const float* W2   = (const float*)d_in[4];
    const float* b2   = (const float*)d_in[5];
    float* out = (float*)d_out;

    const int  N  = in_sizes[0] / DFEAT;     // 200000
    const int  E  = in_sizes[1] / 2;         // 600000
    const long NB = (long)N * DFEAT;         // node-feature element count
    const int* esrc = eidx;
    const int* edst = eidx + E;

    // workspace layout (bytes), 256-aligned slabs
    auto align256 = [](size_t v) { return (v + 255) & ~(size_t)255; };
    char* ws = (char*)d_ws;
    size_t off = 0;
    bf16_t* xb   = (bf16_t*)(ws + off); off = align256(off + (size_t)NB * sizeof(bf16_t));
    float*  h    = (float*) (ws + off); off = align256(off + (size_t)NB * sizeof(float));
    float*  agg  = (float*) (ws + off); off = align256(off + (size_t)NB * sizeof(float));
    float*  dinv = (float*) (ws + off); off = align256(off + (size_t)N  * sizeof(float));
    bf16_t* W1T  = (bf16_t*)(ws + off); off = align256(off + (size_t)DFEAT * DFEAT * sizeof(bf16_t));
    bf16_t* W2T  = (bf16_t*)(ws + off);

    const int T = 256;
    const int blkNB = (int)((NB + T - 1) / T);
    const int blkN  = (N + T - 1) / T;
    const int blkE  = (E + T - 1) / T;
    const int blkEW = (E + 7) / 8;           // 8 waves per block, 1 edge per wave
    const int mtiles = N / 16;               // 12500 (N divisible by 16)

    // prep: bf16 activations + transposed bf16 weights + norm coefficients
    k_cvt_bf16   <<<blkNB, T, 0, stream>>>(x, xb, NB);
    k_transpose_w<<<(DFEAT * DFEAT) / T, T, 0, stream>>>(W1, W2, W1T, W2T);
    k_fill_zero  <<<blkN, T, 0, stream>>>(dinv, N);
    k_degree     <<<blkE, T, 0, stream>>>(edst, dinv, E);
    k_dinv       <<<blkN, T, 0, stream>>>(dinv, N);

    // ---- layer 1 ----
    k_gemm_wmma_bf16<<<mtiles, T, 0, stream>>>(xb, W1T, h);
    k_init_agg      <<<blkNB, T, 0, stream>>>(h, dinv, b1, agg, NB);
    k_scatter       <<<blkEW, T, 0, stream>>>(h, esrc, edst, dinv, agg, E);

    // ReLU + convert to bf16 (reuse xb; layer-1 GEMM input no longer needed)
    k_relu_cvt_bf16 <<<blkNB, T, 0, stream>>>(agg, xb, NB);

    // ---- layer 2 ----
    k_gemm_wmma_bf16<<<mtiles, T, 0, stream>>>(xb, W2T, h);
    k_init_agg      <<<blkNB, T, 0, stream>>>(h, dinv, b2, out, NB);
    k_scatter       <<<blkEW, T, 0, stream>>>(h, esrc, edst, dinv, out, E);
}
